// DAINModel_81088982548605
// MI455X (gfx1250) — compile-verified, matrix operand order checked by reading, not run on hardware
//
#include <hip/hip_runtime.h>
#include <math.h>

// ---------------------------------------------------------------------------
// DAIN forward on MI455X (gfx1250, wave32).
// Implicit-GEMM conv via v_wmma_f32_16x16x32_f16 (8 WMMAs/wave/stage,
// 32 cout x 128 px block tile). Weight tiles double-buffered in LDS via the
// Tensor Data Mover (tensor_load_to_lds, LDS row padding via D# pad fields,
// drained with s_wait_tensorcnt) overlapping DMA with the im2col fill+WMMA.
// ---------------------------------------------------------------------------

typedef __attribute__((ext_vector_type(16))) _Float16 v16h;
typedef __attribute__((ext_vector_type(8)))  _Float16 v8h;
typedef __attribute__((ext_vector_type(8)))  float    v8f;
typedef unsigned int u32;
typedef unsigned int u32x4 __attribute__((ext_vector_type(4)));
typedef int          i32x4 __attribute__((ext_vector_type(4)));
typedef int          i32x8 __attribute__((ext_vector_type(8)));

#if __has_builtin(__builtin_amdgcn_tensor_load_to_lds)
#define HAVE_TDM 1
#else
#define HAVE_TDM 0
#endif
#if defined(__gfx1250__) && !HAVE_TDM
#warning "gfx1250 device pass: __builtin_amdgcn_tensor_load_to_lds NOT available - manual A-tile path in use"
#endif
#if defined(__gfx1250__) && !__has_builtin(__builtin_amdgcn_s_wait_tensorcnt)
#warning "gfx1250 device pass: __builtin_amdgcn_s_wait_tensorcnt NOT available"
#endif

#define ACT_NONE 0
#define ACT_RELU 1
#define ACT_SIGM 2

// LDS row strides (padded to spread banks; keep 32B alignment for v8f/v8h)
#define A_STRIDE 72   // floats per A row (64 data + 8 pad)
#define B_STRIDE 72   // halfs  per B row (64 data + 8 pad)

struct ConvDesc {
  const float* in; const float* w; const float* bias; float* out;
  int Cin, Cout, Hin, Win, Hout, Wout, KH, KW, stride, pad;
  int mode;   // 0 = conv, 1 = transposed-conv (gather form, weights (Cin,Cout,k,k))
  int act;    // ACT_*
  int lstride;// log2(stride) for mode 1 (strides are powers of two here)
};

__device__ __forceinline__ v16h mk_a_from_f32(const float* ra, int off) {
  const v8f lo = *(const v8f*)(ra + off);
  const v8f hi = *(const v8f*)(ra + 16 + off);
  v16h A;
#pragma unroll
  for (int j = 0; j < 8; ++j) { A[j] = (_Float16)lo[j]; A[j + 8] = (_Float16)hi[j]; }
  return A;
}

__device__ __forceinline__ v16h mk_b_from_f16(const _Float16* rb, int off) {
  const v8h lo = *(const v8h*)(rb + off);
  const v8h hi = *(const v8h*)(rb + 16 + off);
  v16h B;
#pragma unroll
  for (int j = 0; j < 8; ++j) { B[j] = lo[j]; B[j + 8] = hi[j]; }
  return B;
}

#if HAVE_TDM
// Issue one 2D TDM: tile 64 x rows f32, row stride Ktot elems, into LDS with
// +8-dword padding after every 64 dwords (-> A_STRIDE rows).
__device__ __forceinline__ void tdm_load_a(const float* gsrc, unsigned ldsaddr,
                                           int remK, int remM, int Ktot) {
  u32x4 g0;
  const unsigned long long ga = (unsigned long long)(const void*)gsrc;
  g0[0] = 1u;                                               // count=1
  g0[1] = ldsaddr;                                          // lds_addr (bytes)
  g0[2] = (u32)(ga & 0xffffffffull);                        // global_addr[31:0]
  g0[3] = (u32)((ga >> 32) & 0x01ffffffull) | 0x80000000u;  // [56:32] | type=2
  i32x8 g1;
  const u32 td0 = (u32)remK, td1 = (u32)remM;
  g1[0] = (int)((2u << 16) | (1u << 20) | (5u << 22) | (7u << 25));
  //            data=4B      pad_en       ivl=64dw     amt=8dw
  g1[1] = (int)((td0 & 0xffffu) << 16);                     // tensor_dim0[15:0]
  g1[2] = (int)((td0 >> 16) | ((td1 & 0xffffu) << 16));     // dim0 hi | dim1 lo
  g1[3] = (int)((td1 >> 16) | (64u << 16));                 // dim1 hi | tile_dim0=64
  g1[4] = 32;                                               // tile_dim1=32
  g1[5] = (int)(u32)Ktot;                                   // dim0_stride lo32
  g1[6] = 0; g1[7] = 0;
  const i32x4 gz = { 0, 0, 0, 0 };
#if defined(__clang_major__) && __clang_major__ >= 23
  const i32x8 gz8 = { 0, 0, 0, 0, 0, 0, 0, 0 };
  __builtin_amdgcn_tensor_load_to_lds(g0, g1, gz, gz, gz8, 0);
#else
  __builtin_amdgcn_tensor_load_to_lds(g0, g1, gz, gz, 0);
#endif
}
#endif

// Block = 128 threads = 4 waves. Block tile: 32 couts x 128 pixels.
// Per K-stage (K=64): each wave runs 8 v_wmma_f32_16x16x32_f16
// (2 M-subtiles x 2 N-subtiles x 2 K-chunks).
__global__ __launch_bounds__(128) void conv_wmma(ConvDesc d) {
  __shared__ __align__(32) float    sA[2][32 * A_STRIDE];  // weights, double buffer
  __shared__ __align__(32) _Float16 sB[128 * B_STRIDE];    // im2col^T [pix][k]

  const int tid  = threadIdx.x;
  const int wv   = tid >> 5;
  const int lane = tid & 31;
  const int cout0 = blockIdx.y * 32;
  const int Npix  = d.Hout * d.Wout;
  const int pix0  = blockIdx.x * 128;
  const int khw   = d.KH * d.KW;
  const int Ktot  = d.Cin * khw;
  const int HinWin = d.Hin * d.Win;
  const int nstages = (Ktot + 63) >> 6;

  // B fill: one pixel per thread, decoded once for the whole kernel
  const int pF  = pix0 + tid;
  const int oyF = pF / d.Wout;
  const int oxF = pF - oyF * d.Wout;
  const int oys = oyF * d.stride - d.pad;   // mode 0 base coords
  const int oxs = oxF * d.stride - d.pad;
  const int typ = oyF + d.pad;              // mode 1 base coords
  const int txp = oxF + d.pad;
  const bool pValid = (pF < Npix);
  _Float16* sbRow = &sB[(size_t)tid * B_STRIDE];

  // A manual-fill lanes: kk fixed per thread, rows step by 2
  const int kkA = tid & 63;
  const int mA0 = tid >> 6;

  const int khalf8 = (lane >> 4) << 3;   // 0 or 8: K half-set owned by this lane
  const int mr     = lane & 15;

  v8f acc[4] = { {}, {}, {}, {} };

#if HAVE_TDM
  const bool use_tdm = (d.mode == 0);
  if (use_tdm && wv == 0) {
    tdm_load_a(d.w + (size_t)cout0 * Ktot,
               (unsigned)(unsigned long long)(&sA[0][0]),
               Ktot, d.Cout - cout0, Ktot);
  }
#endif

  for (int s = 0; s < nstages; ++s) {
    const int kb = s << 6;
    const int buf = s & 1;

#if HAVE_TDM
    if (use_tdm) {
      if (wv == 0 && (s + 1) < nstages) {
        tdm_load_a(d.w + (size_t)cout0 * Ktot + (kb + 64),
                   (unsigned)(unsigned long long)(&sA[buf ^ 1][0]),
                   Ktot - (kb + 64), d.Cout - cout0, Ktot);
      }
    } else
#endif
    {
      // manual A fill (deconv, or no TDM builtin): coalesced along kk
      const int k = kb + kkA;
      const bool kval = (k < Ktot);
      if (d.mode == 0) {
        for (int m = mA0; m < 32; m += 2) {
          const int cout = cout0 + m;
          float v = 0.0f;
          if (kval && cout < d.Cout) v = d.w[(size_t)cout * Ktot + k];
          sA[buf][m * A_STRIDE + kkA] = v;
        }
      } else {
        int cin = 0, r = 0;
        if (kval) { cin = k / khw; r = k - cin * khw; }
        const float* wb = d.w + (size_t)cin * d.Cout * khw + r;
        for (int m = mA0; m < 32; m += 2) {
          const int cout = cout0 + m;
          float v = 0.0f;
          if (kval && cout < d.Cout) v = wb[(size_t)cout * khw];
          sA[buf][m * A_STRIDE + kkA] = v;
        }
      }
    }

    // ---- B fill: 64 taps for this thread's pixel, incremental decode -----
    {
      int cin = kb / khw;
      int r   = kb - cin * khw;
      int ry  = r / d.KW;
      int rx  = r - ry * d.KW;
      const float* inC = d.in + (size_t)cin * HinWin;
      if ((kb + 64) < Ktot)
        __builtin_prefetch(d.in + (size_t)((kb + 64) / khw) * HinWin + (tid & 63), 0, 0);
      for (int kk = 0; kk < 64; ++kk) {
        _Float16 v = (_Float16)0.0f;
        const bool kval = (kb + kk) < Ktot;
        if (pValid && kval) {
          if (d.mode == 0) {
            const int iy = oys + ry, ix = oxs + rx;
            if (iy >= 0 && iy < d.Hin && ix >= 0 && ix < d.Win)
              v = (_Float16)inC[iy * d.Win + ix];
          } else {
            const int ty = typ - ry, tx = txp - rx;
            const int ms = d.stride - 1;
            if (ty >= 0 && tx >= 0 && ((ty & ms) | (tx & ms)) == 0) {
              const int iy = ty >> d.lstride, ix = tx >> d.lstride;
              if (iy < d.Hin && ix < d.Win) v = (_Float16)inC[iy * d.Win + ix];
            }
          }
        }
        sbRow[kk] = v;
        if (++rx == d.KW) {
          rx = 0;
          if (++ry == d.KH) { ry = 0; ++cin; inC += HinWin; }
        }
      }
    }

#if HAVE_TDM && __has_builtin(__builtin_amdgcn_s_wait_tensorcnt)
    if (use_tdm) {
      if ((s + 1) < nstages) __builtin_amdgcn_s_wait_tensorcnt(1);
      else                   __builtin_amdgcn_s_wait_tensorcnt(0);
    }
#endif
    __syncthreads();

    // ---- 8 WMMAs per wave: 2 K-chunks x (2 M x 2 N) ----------------------
    const float*    sAc = &sA[buf][0];
    const _Float16* sBc = &sB[(size_t)(wv * 32) * B_STRIDE];
#pragma unroll
    for (int kc = 0; kc < 2; ++kc) {
      const v16h A0 = mk_a_from_f32(sAc + (mr)      * A_STRIDE + kc * 32, khalf8);
      const v16h A1 = mk_a_from_f32(sAc + (16 + mr) * A_STRIDE + kc * 32, khalf8);
      const v16h B0 = mk_b_from_f16(sBc + (mr)      * B_STRIDE + kc * 32, khalf8);
      const v16h B1 = mk_b_from_f16(sBc + (16 + mr) * B_STRIDE + kc * 32, khalf8);
      acc[0] = __builtin_amdgcn_wmma_f32_16x16x32_f16(false, A0, false, B0, (short)0, acc[0], false, false);
      acc[1] = __builtin_amdgcn_wmma_f32_16x16x32_f16(false, A0, false, B1, (short)0, acc[1], false, false);
      acc[2] = __builtin_amdgcn_wmma_f32_16x16x32_f16(false, A1, false, B0, (short)0, acc[2], false, false);
      acc[3] = __builtin_amdgcn_wmma_f32_16x16x32_f16(false, A1, false, B1, (short)0, acc[3], false, false);
    }
    __syncthreads();
  }

  // ---- epilogue: D layout VGPR r -> M = r + 8*(lane>=16), N = lane&15 ----
  const int nlow = lane & 15;
#pragma unroll
  for (int mt = 0; mt < 2; ++mt) {
#pragma unroll
    for (int nt = 0; nt < 2; ++nt) {
      const v8f a = acc[mt * 2 + nt];
      const int p = pix0 + wv * 32 + nt * 16 + nlow;
#pragma unroll
      for (int r = 0; r < 8; ++r) {
        const int m = mt * 16 + r + ((lane >> 4) << 3);
        const int cout = cout0 + m;
        if (cout < d.Cout && p < Npix) {
          float v = a[r] + d.bias[cout];
          if (d.act == ACT_RELU)      v = v > 0.0f ? v : 0.0f;
          else if (d.act == ACT_SIGM) v = 1.0f / (1.0f + __expf(-v));
          d.out[(size_t)cout * Npix + p] = v;
        }
      }
    }
  }
}

// ---------------------------------------------------------------------------
// Elementwise / memory-bound helper kernels
// ---------------------------------------------------------------------------

__global__ void k_copy(const float* __restrict__ s, float* __restrict__ dst, int n) {
  const int i = blockIdx.x * 256 + threadIdx.x;
  if (i < n) dst[i] = s[i];
}

__global__ void k_maxpool2(const float* __restrict__ in, float* __restrict__ out,
                           int C, int H, int W) {
  const int Ho = H >> 1, Wo = W >> 1;
  const int n = C * Ho * Wo;
  const int i = blockIdx.x * 256 + threadIdx.x;
  if (i >= n) return;
  const int c = i / (Ho * Wo), r = i - c * (Ho * Wo);
  const int y = r / Wo, x = r - y * Wo;
  const float* p = &in[((size_t)c * H + 2 * y) * W + 2 * x];
  out[i] = fmaxf(fmaxf(p[0], p[1]), fmaxf(p[W], p[W + 1]));
}

__global__ void k_resize_bilinear(const float* __restrict__ in, float* __restrict__ out,
                                  int C, int Hin, int Win, int Hout, int Wout) {
  const int n = C * Hout * Wout;
  const int i = blockIdx.x * 256 + threadIdx.x;
  if (i >= n) return;
  const int c = i / (Hout * Wout), r = i - c * (Hout * Wout);
  const int y = r / Wout, x = r - y * Wout;
  float sy = (y + 0.5f) * (float)Hin / (float)Hout - 0.5f;
  float sx = (x + 0.5f) * (float)Win / (float)Wout - 0.5f;
  sy = fminf(fmaxf(sy, 0.0f), (float)(Hin - 1));
  sx = fminf(fmaxf(sx, 0.0f), (float)(Win - 1));
  const int y0 = (int)sy, x0 = (int)sx;
  const int y1 = min(y0 + 1, Hin - 1), x1 = min(x0 + 1, Win - 1);
  const float wy = sy - y0, wx = sx - x0;
  const float* b = &in[(size_t)c * Hin * Win];
  out[i] = b[y0 * Win + x0] * (1 - wx) * (1 - wy) + b[y0 * Win + x1] * wx * (1 - wy)
         + b[y1 * Win + x0] * (1 - wx) * wy       + b[y1 * Win + x1] * wx * wy;
}

__global__ void k_scale2(const float* __restrict__ f, float* __restrict__ a,
                         float* __restrict__ b, int n, float s1, float s2) {
  const int i = blockIdx.x * 256 + threadIdx.x;
  if (i < n) { const float v = f[i]; a[i] = v * s1; b[i] = v * s2; }
}

// bilinear, border clamp, align_corners=True, pixel-unit flow
__global__ void k_grid_sample(const float* __restrict__ img, const float* __restrict__ flow,
                              float* __restrict__ out, int C, int H, int W) {
  const int HW = H * W;
  const int p = blockIdx.x * 256 + threadIdx.x;
  if (p >= HW) return;
  const int y = p / W, x = p - y * W;
  float cx = fminf(fmaxf((float)x + flow[p],      0.0f), (float)(W - 1));
  float cy = fminf(fmaxf((float)y + flow[HW + p], 0.0f), (float)(H - 1));
  const int x0 = (int)cx, y0 = (int)cy;
  const int x1 = min(x0 + 1, W - 1), y1 = min(y0 + 1, H - 1);
  const float wx = cx - x0, wy = cy - y0;
  for (int c = 0; c < C; ++c) {
    const float* b = &img[(size_t)c * HW];
    out[(size_t)c * HW + p] =
        b[y0 * W + x0] * (1 - wx) * (1 - wy) + b[y0 * W + x1] * wx * (1 - wy)
      + b[y1 * W + x0] * (1 - wx) * wy       + b[y1 * W + x1] * wx * wy;
  }
}

__global__ void k_softmax_ch(float* __restrict__ data, int Cch, int Npix) {
  const int p = blockIdx.x * 256 + threadIdx.x;
  if (p >= Npix) return;
  float mx = -1e30f;
  for (int c = 0; c < Cch; ++c) mx = fmaxf(mx, data[(size_t)c * Npix + p]);
  float s = 0.0f;
  for (int c = 0; c < Cch; ++c) s += __expf(data[(size_t)c * Npix + p] - mx);
  const float inv = 1.0f / s;
  for (int c = 0; c < Cch; ++c) {
    const size_t idx = (size_t)c * Npix + p;
    data[idx] = __expf(data[idx] - mx) * inv;
  }
}

// 51x51 adaptive conv: out[c,p] = sum_k kern[k,p] * img_padded[c, y+dy, x+dx]
__global__ void k_adaptive(const float* __restrict__ img, const float* __restrict__ kern,
                           float* __restrict__ out, int H, int W, int K) {
  const int HW = H * W;
  const int p = blockIdx.x * 256 + threadIdx.x;
  if (p >= HW) return;
  const int y = p / W, x = p - y * W;
  const int pad = K / 2;
  float a0 = 0.f, a1 = 0.f, a2 = 0.f;
  for (int dy = 0; dy < K; ++dy) {
    const int iy = y + dy - pad;
    const bool okY = (iy >= 0 && iy < H);
    for (int dx = 0; dx < K; ++dx) {
      const float kv = kern[(size_t)(dy * K + dx) * HW + p];
      const int ix = x + dx - pad;
      if (okY && ix >= 0 && ix < W) {
        const int q = iy * W + ix;
        a0 += kv * img[q];
        a1 += kv * img[HW + q];
        a2 += kv * img[2 * HW + q];
      }
    }
  }
  out[p] = a0; out[HW + p] = a1; out[2 * HW + p] = a2;
}

__global__ void k_blend(const float* __restrict__ wd1, const float* __restrict__ wd2,
                        const float* __restrict__ a1, const float* __restrict__ a2,
                        float* __restrict__ occ, float* __restrict__ alpha,
                        float* __restrict__ interp, int HW) {
  const int p = blockIdx.x * 256 + threadIdx.x;
  if (p >= HW) return;
  const float dd = wd1[p] - wd2[p];
  const float o  = fabsf(dd) > 0.1f ? 1.0f : 0.0f;
  float al = fminf(fmaxf(0.5f + 0.3f * dd, 0.0f), 1.0f);
  al = al * (1.0f - o) + 0.5f * o;
  occ[p] = o; alpha[p] = al;
  for (int c = 0; c < 3; ++c)
    interp[(size_t)c * HW + p] = al * a1[(size_t)c * HW + p] + (1.0f - al) * a2[(size_t)c * HW + p];
}

// ---------------------------------------------------------------------------
// Host orchestration
// ---------------------------------------------------------------------------

static int ilog2(int v) { int l = 0; while ((1 << l) < v) ++l; return l; }

static void launch_conv(hipStream_t s, const float* in, const float* w, const float* b,
                        float* out, int Cin, int Cout, int Hin, int Win,
                        int KH, int KW, int stride, int pad, int mode, int act) {
  int Hout, Wout;
  if (mode == 0) { Hout = (Hin + 2 * pad - KH) / stride + 1; Wout = (Win + 2 * pad - KW) / stride + 1; }
  else           { Hout = (Hin - 1) * stride - 2 * pad + KH; Wout = (Win - 1) * stride - 2 * pad + KW; }
  ConvDesc d{in, w, b, out, Cin, Cout, Hin, Win, Hout, Wout, KH, KW, stride, pad,
             mode, act, ilog2(stride)};
  dim3 grid((Hout * Wout + 127) / 128, (Cout + 31) / 32);
  conv_wmma<<<grid, dim3(128), 0, s>>>(d);
}

static inline dim3 g1(int n) { return dim3((n + 255) / 256); }

extern "C" void kernel_launch(void* const* d_in, const int* in_sizes, int n_in,
                              void* d_out, int out_size, void* d_ws, size_t ws_size,
                              hipStream_t stream) {
  (void)in_sizes; (void)n_in; (void)out_size; (void)ws_size;
  const float* frame1 = (const float*)d_in[0];
  const float* frame2 = (const float*)d_in[1];
  const float *dw1 = (const float*)d_in[2],  *db1 = (const float*)d_in[3];
  const float *dw2 = (const float*)d_in[4],  *db2 = (const float*)d_in[5];
  const float *dw3 = (const float*)d_in[6],  *db3 = (const float*)d_in[7];
  const float *dw4 = (const float*)d_in[8],  *db4 = (const float*)d_in[9];
  const float *dw5 = (const float*)d_in[10], *db5 = (const float*)d_in[11];
  const float *dt1w = (const float*)d_in[12], *dt1b = (const float*)d_in[13];
  const float *dt2w = (const float*)d_in[14], *dt2b = (const float*)d_in[15];
  const float *dt3w = (const float*)d_in[16], *dt3b = (const float*)d_in[17];
  const float *dt4w = (const float*)d_in[18], *dt4b = (const float*)d_in[19];
  const float *p0w1 = (const float*)d_in[20], *p0b1 = (const float*)d_in[21];
  const float *p0w2 = (const float*)d_in[22], *p0b2 = (const float*)d_in[23];
  const float *h0w1 = (const float*)d_in[24], *h0b1 = (const float*)d_in[25];
  const float *h0w2 = (const float*)d_in[26], *h0b2 = (const float*)d_in[27];
  const float *h0w3 = (const float*)d_in[28], *h0b3 = (const float*)d_in[29];
  const float *k1w  = (const float*)d_in[30], *k1b  = (const float*)d_in[31];
  const float *k2w  = (const float*)d_in[32], *k2b  = (const float*)d_in[33];
  const float *k3w  = (const float*)d_in[34], *k3b  = (const float*)d_in[35];
  const float *k4w  = (const float*)d_in[36], *k4b  = (const float*)d_in[37];

  float* ws  = (float*)d_ws;
  float* out = (float*)d_out;
  const int HW = 128 * 128;

  // d_out packing (tuple return order): interp, depth1, depth2, flow, occ, alpha
  float* o_interp = out;
  float* o_d1     = out + 3 * HW;
  float* o_d2     = out + 4 * HW;
  float* o_flow   = out + 5 * HW;
  float* o_occ    = out + 7 * HW;
  float* o_alpha  = out + 8 * HW;

  // static workspace layout (floats); peak ~248 MB dominated by the 2601-ch map
  const size_t T0 = 0,        T1 = 262144,  T2 = 393216,  T3 = 458752,  T4 = 491520;
  const size_t T5 = 507904,   T6 = 540672,  T7 = 606208;
  const size_t CAT6 = 737280, F1 = 835584,  F2 = 1884160, FP = 2932736;
  const size_t COMB = 3194880, H1 = 3719168, H2 = 3981312, FLOW0 = 4112384;
  const size_t F1T = 4120576, F2T = 4153344, WARP1 = 4186112, WARP2 = 4235264;
  const size_t WD1 = 4284416, WD2 = 4300800, KI = 4317184, A1 = 4481024, A2 = 4530176;
  const size_t K1O = 4579328, K2O = 6676480, K3O = 10870784, K4O = 19259392;

  // ---- depth nets (shared temps, sequential) ------------------------------
  auto depth_net = [&](const float* f, float* dep) {
    launch_conv(stream, f,       dw1,  db1,  ws + T0, 3,   64,  128, 128, 7, 7, 2, 3, 0, ACT_RELU);
    launch_conv(stream, ws + T0, dw2,  db2,  ws + T1, 64,  128, 64,  64,  5, 5, 2, 2, 0, ACT_RELU);
    launch_conv(stream, ws + T1, dw3,  db3,  ws + T2, 128, 256, 32,  32,  3, 3, 2, 1, 0, ACT_RELU);
    launch_conv(stream, ws + T2, dw4,  db4,  ws + T3, 256, 512, 16,  16,  3, 3, 2, 1, 0, ACT_RELU);
    launch_conv(stream, ws + T3, dw5,  db5,  ws + T4, 512, 256, 8,   8,   3, 3, 1, 1, 0, ACT_RELU);
    launch_conv(stream, ws + T4, dt1w, dt1b, ws + T5, 256, 128, 8,   8,   4, 4, 2, 1, 1, ACT_NONE);
    launch_conv(stream, ws + T5, dt2w, dt2b, ws + T6, 128, 64,  16,  16,  4, 4, 2, 1, 1, ACT_NONE);
    launch_conv(stream, ws + T6, dt3w, dt3b, ws + T7, 64,  32,  32,  32,  4, 4, 2, 1, 1, ACT_NONE);
    launch_conv(stream, ws + T7, dt4w, dt4b, dep,     32,  1,   64,  64,  4, 4, 2, 1, 1, ACT_SIGM);
  };
  depth_net(frame1, o_d1);
  depth_net(frame2, o_d2);

  // ---- flow branch --------------------------------------------------------
  k_copy<<<g1(3 * HW), 256, 0, stream>>>(frame1, ws + CAT6, 3 * HW);
  k_copy<<<g1(3 * HW), 256, 0, stream>>>(frame2, ws + CAT6 + 3 * HW, 3 * HW);
  launch_conv(stream, ws + CAT6, p0w1, p0b1, ws + F1, 6,  64, 128, 128, 3, 3, 1, 1, 0, ACT_RELU);
  launch_conv(stream, ws + F1,   p0w2, p0b2, ws + F2, 64, 64, 128, 128, 3, 3, 1, 1, 0, ACT_RELU);
  k_maxpool2<<<g1(64 * 64 * 64), 256, 0, stream>>>(ws + F2, ws + FP, 64, 128, 128);
  k_copy<<<g1(64 * 4096), 256, 0, stream>>>(ws + FP, ws + COMB, 64 * 4096);
  k_copy<<<g1(64 * 4096), 256, 0, stream>>>(ws + FP, ws + COMB + 64 * 4096, 64 * 4096);
  launch_conv(stream, ws + COMB, h0w1, h0b1, ws + H1,    128, 64, 64, 64, 3, 3, 1, 1, 0, ACT_RELU);
  launch_conv(stream, ws + H1,   h0w2, h0b2, ws + H2,    64,  32, 64, 64, 3, 3, 1, 1, 0, ACT_RELU);
  launch_conv(stream, ws + H2,   h0w3, h0b3, ws + FLOW0, 32,  2,  64, 64, 3, 3, 1, 1, 0, ACT_NONE);
  k_resize_bilinear<<<g1(2 * HW), 256, 0, stream>>>(ws + FLOW0, o_flow, 2, 64, 64, 128, 128);

  // ---- warps --------------------------------------------------------------
  k_scale2<<<g1(2 * HW), 256, 0, stream>>>(o_flow, ws + F1T, ws + F2T, 2 * HW, 0.5f, -0.5f);
  k_grid_sample<<<g1(HW), 256, 0, stream>>>(frame1, ws + F1T, ws + WARP1, 3, 128, 128);
  k_grid_sample<<<g1(HW), 256, 0, stream>>>(frame2, ws + F2T, ws + WARP2, 3, 128, 128);
  k_grid_sample<<<g1(HW), 256, 0, stream>>>(o_d1,   ws + F1T, ws + WD1,   1, 128, 128);
  k_grid_sample<<<g1(HW), 256, 0, stream>>>(o_d2,   ws + F2T, ws + WD2,   1, 128, 128);

  // ---- kernel estimator (dominant compute) --------------------------------
  k_copy<<<g1(3 * HW), 256, 0, stream>>>(frame1, ws + KI, 3 * HW);
  k_copy<<<g1(3 * HW), 256, 0, stream>>>(frame2, ws + KI + 3 * HW, 3 * HW);
  k_copy<<<g1(2 * HW), 256, 0, stream>>>(ws + F1T, ws + KI + 6 * HW, 2 * HW);
  k_copy<<<g1(2 * HW), 256, 0, stream>>>(ws + F2T, ws + KI + 8 * HW, 2 * HW);
  launch_conv(stream, ws + KI,  k1w, k1b, ws + K1O, 10,  128,  128, 128, 3, 3, 1, 1, 0, ACT_RELU);
  launch_conv(stream, ws + K1O, k2w, k2b, ws + K2O, 128, 256,  128, 128, 3, 3, 1, 1, 0, ACT_RELU);
  launch_conv(stream, ws + K2O, k3w, k3b, ws + K3O, 256, 512,  128, 128, 3, 3, 1, 1, 0, ACT_RELU);
  launch_conv(stream, ws + K3O, k4w, k4b, ws + K4O, 512, 2601, 128, 128, 3, 3, 1, 1, 0, ACT_NONE);
  k_softmax_ch<<<g1(HW), 256, 0, stream>>>(ws + K4O, 2601, HW);

  // ---- adaptive conv + blend ---------------------------------------------
  k_adaptive<<<g1(HW), 256, 0, stream>>>(ws + WARP1, ws + K4O, ws + A1, 128, 128, 51);
  k_adaptive<<<g1(HW), 256, 0, stream>>>(ws + WARP2, ws + K4O, ws + A2, 128, 128, 51);
  k_blend<<<g1(HW), 256, 0, stream>>>(ws + WD1, ws + WD2, ws + A1, ws + A2,
                                      o_occ, o_alpha, o_interp, HW);
}
